// TFPerceiverSelfAttention_9259949490490
// MI455X (gfx1250) — matrix operand order repaired and checked by reading, not compiled, not run on hardware
//
#include <hip/hip_runtime.h>
#include <math.h>

// ---------------------------------------------------------------------------
// TFPerceiverSelfAttention forward for MI455X (gfx1250, wave32, WMMA).
//   B=4 QL=512 KVL=4096 QDIM=1024 KVDIM=768 H=8 DQK=DV=128
// Pipeline: LN->f16, weights->f16 transposed, 3 WMMA GEMM projections,
// flash-attention with v_wmma_f32_16x16x32_f16 for QK^T and PV, with K/V
// tiles staged into LDS via GLOBAL_LOAD_ASYNC_TO_LDS_B128 (ASYNCcnt path)
// and shared by the 4 waves of each block.
// ---------------------------------------------------------------------------

typedef _Float16 half_t;
typedef __attribute__((ext_vector_type(16))) _Float16 v16h;
typedef __attribute__((ext_vector_type(8)))  _Float16 v8h;
typedef __attribute__((ext_vector_type(8)))  float    v8f;

#define BATCH 4
#define QLEN  512
#define KVLEN 4096
#define QDIM  1024
#define KVDIM 768
#define NH    8
#define DH    128
#define QKCH  1024
#define VCH   1024

// ---- WMMA fragment helpers (contiguous k order used by both A and B, wave32) ----------
// A (16x32 f16): lane m=lane&15, k0=16*(lane>>4): halves 0..15 = K k0..k0+15
//                -> contiguous load p[0..15] (matches B fragments)
__device__ __forceinline__ v16h load_a16_gap(const half_t* p) {
  union { v16h v; v8h h[2]; } u;
  u.h[0] = *(const v8h*)(p);
  u.h[1] = *(const v8h*)(p + 8);
  return u.v;
}
// B (32x16 f16): lane n=lane&15, k0=16*(lane>>4): halves 0..15 = K k0..k0+15
//                at column n -> contiguous 32-byte load
__device__ __forceinline__ v16h load_b16_contig(const half_t* p) {
  union { v16h v; v8h h[2]; } u;
  u.h[0] = *(const v8h*)(p);
  u.h[1] = *(const v8h*)(p + 8);
  return u.v;
}
__device__ __forceinline__ v8f wmma_f16(v16h a, v16h b, v8f c) {
  return __builtin_amdgcn_wmma_f32_16x16x32_f16(
      /*neg_a=*/false, a, /*neg_b=*/false, b,
      /*c_mod=*/(short)0, c, /*reuse_a=*/false, /*reuse_b=*/false);
}

// ---- async 16-byte global -> LDS copy (CDNA5 §10.7 / §15.18, ASYNCcnt) -----
// VDST = LDS byte address (low 32 bits of the generic shared pointer; the LDS
// aperture lives in bits [63:32]), VADDR = 64-bit global address, GV mode.
__device__ __forceinline__ void async_copy_b128(const half_t* g, half_t* l) {
  unsigned loff = (unsigned)(size_t)l;
  asm volatile("global_load_async_to_lds_b128 %0, %1, off"
               :: "v"(loff), "v"(g) : "memory");
}
__device__ __forceinline__ void wait_asynccnt0() {
  asm volatile("s_wait_asynccnt 0" ::: "memory");
}

// ---------------------------------------------------------------------------
// Kernel 1: LayerNorm (f32 in) -> f16 out.  One 256-thread block per row.
// ---------------------------------------------------------------------------
template <int WIDTH>
__global__ void ln_to_f16(const float* __restrict__ x,
                          const float* __restrict__ g,
                          const float* __restrict__ b,
                          half_t* __restrict__ y) {
  __shared__ float s1[256];
  __shared__ float s2[256];
  const int row = blockIdx.x;
  const float* xr = x + (size_t)row * WIDTH;
  float a = 0.f, sq = 0.f;
  for (int i = threadIdx.x; i < WIDTH; i += 256) {
    float v = xr[i];
    a += v; sq += v * v;
  }
  s1[threadIdx.x] = a; s2[threadIdx.x] = sq;
  __syncthreads();
  for (int off = 128; off > 0; off >>= 1) {
    if ((int)threadIdx.x < off) {
      s1[threadIdx.x] += s1[threadIdx.x + off];
      s2[threadIdx.x] += s2[threadIdx.x + off];
    }
    __syncthreads();
  }
  const float mean = s1[0] * (1.0f / WIDTH);
  const float var  = s2[0] * (1.0f / WIDTH) - mean * mean;
  const float inv  = rsqrtf(var + 1e-3f);
  half_t* yr = y + (size_t)row * WIDTH;
  for (int i = threadIdx.x; i < WIDTH; i += 256)
    yr[i] = (half_t)((xr[i] - mean) * inv * g[i] + b[i]);
}

// ---------------------------------------------------------------------------
// Kernel 2: weight transpose + f32->f16.  WT[n][k] = W[k][n].
// ---------------------------------------------------------------------------
__global__ void wtrans_f16(const float* __restrict__ W, half_t* __restrict__ WT,
                           int Kd, int N) {
  size_t i = (size_t)blockIdx.x * blockDim.x + threadIdx.x;
  if (i >= (size_t)Kd * N) return;
  int n = (int)(i / Kd);
  int k = (int)(i % Kd);
  WT[i] = (half_t)W[(size_t)k * N + n];
}

// ---------------------------------------------------------------------------
// Kernel 3: projection GEMM  C = A[M,Kd] @ W[Kd,N] + bias, via WT[N,Kd] f16.
// One wave per 16x16 output tile; K-loop in steps of 32 -> 1 WMMA/step.
// mode 0: Q  -> [B,H,QL,128] f16, pre-scaled by 1/sqrt(128)
// mode 1: K  -> [B,H,KVL,128] f16
// mode 2: V  -> [B,H,128,KVL] f16 (transposed for PV B-fragments)
// ---------------------------------------------------------------------------
__global__ void proj_gemm(const half_t* __restrict__ A,
                          const half_t* __restrict__ WT,
                          const float* __restrict__ bias,
                          half_t* __restrict__ out,
                          int Mtot, int N, int Kd, int rowsPerBatch,
                          int mode, float scale) {
  const int gwave = (int)(((size_t)blockIdx.x * blockDim.x + threadIdx.x) >> 5);
  const int lane  = threadIdx.x & 31;
  const int tilesN = N >> 4;
  const int totalTiles = (Mtot >> 4) * tilesN;
  if (gwave >= totalTiles) return;           // wave-uniform: EXEC stays all-1s
  const int tileM = gwave / tilesN;
  const int tileN = gwave % tilesN;
  const int half16 = lane >> 4;
  const int nlo    = lane & 15;

  const half_t* arow = A  + (size_t)(tileM * 16 + nlo) * Kd + half16 * 16;
  const half_t* brow = WT + (size_t)(tileN * 16 + nlo) * Kd + half16 * 16;

  v8f acc = {};
  for (int k = 0; k < Kd; k += 32) {
    v16h af = load_a16_gap(arow + k);
    v16h bf = load_b16_contig(brow + k);
    acc = wmma_f16(af, bf, acc);
  }

  const int col = tileN * 16 + nlo;
  const int h = col >> 7;            // head
  const int d = col & 127;           // dim within head
  const float bb = bias[col];
  for (int r = 0; r < 8; r++) {
    const int row  = tileM * 16 + r + 8 * half16;   // C/D layout: M = r + 8*(lane>=16)
    const int bidx = row / rowsPerBatch;
    const int pos  = row % rowsPerBatch;
    const float v  = (acc[r] + bb) * scale;
    size_t o;
    if (mode == 2)
      o = (((size_t)bidx * NH + h) * DH + d) * (size_t)rowsPerBatch + pos;
    else
      o = (((size_t)bidx * NH + h) * (size_t)rowsPerBatch + pos) * DH + d;
    out[o] = (half_t)v;
  }
}

// ---------------------------------------------------------------------------
// Kernel 4: flash attention.  4 waves / block, all on the same (b,h), each
// owning one 16-row q-tile.  Per 32-key step the block async-stages the
// K chunk (32x128) and V^T chunk (128x32) into LDS (GLOBAL_LOAD_ASYNC_TO_LDS,
// s_wait_asynccnt), then every wave runs 8 WMMA for S(16x32), an online
// softmax (shfl_xor within the 16-lane half that owns each row), P -> LDS ->
// A-fragment, and 8 WMMA for PV.  Context accumulator: 16x128 f32 per wave.
// ---------------------------------------------------------------------------
__global__ void flash_attn(const half_t* __restrict__ Qf,
                           const half_t* __restrict__ Kf,
                           const half_t* __restrict__ Vt,
                           float* __restrict__ out) {
  __shared__ __align__(16) half_t k_tile[32][DH];     // 8 KB: K[kv..kv+31][0..127]
  __shared__ __align__(16) half_t v_tile[DH][32];     // 8 KB: V^T[dv][kv..kv+31]
  __shared__ __align__(16) half_t plds_all[4][16][32];// 1 KB per wave: P staging

  const int lane   = threadIdx.x & 31;
  const int wid    = threadIdx.x >> 5;                // wave in block (0..3)
  const int tid    = threadIdx.x;
  const int bh     = blockIdx.x >> 3;                 // 8 blocks per (b,h)
  const int qt     = (blockIdx.x & 7) * 4 + wid;      // q-tile for this wave
  if (bh >= BATCH * NH) return;                       // wave-uniform guard

  half_t (*plds)[32] = plds_all[wid];
  const half_t* Qbh = Qf + (size_t)bh * QLEN  * DH;
  const half_t* Kbh = Kf + (size_t)bh * KVLEN * DH;
  const half_t* Vbh = Vt + (size_t)bh * DH * KVLEN;

  const int half16 = lane >> 4;
  const int nlo    = lane & 15;

  // Q A-fragments for the whole 128-wide head dim (4 x 16x32)
  const half_t* qrow = Qbh + (size_t)(qt * 16 + nlo) * DH + half16 * 16;
  v16h qfrag[4];
  for (int t = 0; t < 4; t++) qfrag[t] = load_a16_gap(qrow + t * 32);

  float m_run[8], l_run[8];
  v8f acc[8] = {};
  for (int r = 0; r < 8; r++) { m_run[r] = -3.0e38f; l_run[r] = 0.f; }

  for (int kv = 0; kv < KVLEN; kv += 32) {
    // -- make sure all waves finished reading the previous tiles --
    __syncthreads();
    // -- async-stage K chunk: 32 rows x 256 B = 512 x b128, 4 per thread --
    for (int i = tid; i < 512; i += 128) {
      const int row = i >> 4, c = i & 15;             // c: 16-byte chunk in row
      async_copy_b128(Kbh + (size_t)(kv + row) * DH + c * 8, &k_tile[row][c * 8]);
    }
    // -- async-stage V^T chunk: 128 rows x 64 B = 512 x b128, 4 per thread --
    for (int i = tid; i < 512; i += 128) {
      const int row = i >> 2, c = i & 3;
      async_copy_b128(Vbh + (size_t)row * KVLEN + kv + c * 8, &v_tile[row][c * 8]);
    }
    wait_asynccnt0();        // this wave's async writes to LDS are done
    __syncthreads();         // every wave's writes are visible block-wide

    // ---- S = (Q/sqrt(d)) K^T : two 16x16 tiles, f32 accumulate ----
    v8f s0 = {}, s1 = {};
    for (int t = 0; t < 4; t++)
      s0 = wmma_f16(qfrag[t], load_b16_contig(&k_tile[nlo][t * 32 + half16 * 16]), s0);
    for (int t = 0; t < 4; t++)
      s1 = wmma_f16(qfrag[t], load_b16_contig(&k_tile[16 + nlo][t * 32 + half16 * 16]), s1);

    // ---- online softmax; row (r + 8*half16) lives in one 16-lane half ----
    for (int r = 0; r < 8; r++) {
      float a0 = s0[r], a1 = s1[r];
      float mx = fmaxf(a0, a1);
      for (int off = 1; off < 16; off <<= 1) mx = fmaxf(mx, __shfl_xor(mx, off, 32));
      const float m_new = fmaxf(m_run[r], mx);
      const float corr  = __expf(m_run[r] - m_new);
      m_run[r] = m_new;
      const float p0 = __expf(a0 - m_new);
      const float p1 = __expf(a1 - m_new);
      float ps = p0 + p1;
      for (int off = 1; off < 16; off <<= 1) ps += __shfl_xor(ps, off, 32);
      l_run[r] = l_run[r] * corr + ps;
      for (int t = 0; t < 8; t++) acc[t][r] *= corr;
      // stage P (D-layout) into wave-private LDS row-major [m][kv32]
      plds[r + 8 * half16][nlo]      = (half_t)p0;
      plds[r + 8 * half16][16 + nlo] = (half_t)p1;
    }
    // LDS stores from this wave are in-order; make the dependency explicit.
    asm volatile("s_wait_dscnt 0" ::: "memory");

    // ---- ctx += P(16x32) @ V(32x128): one A-frag, 8 B-frags/WMMAs ----
    const v16h pf = load_a16_gap(&plds[nlo][half16 * 16]);
    for (int t = 0; t < 8; t++)
      acc[t] = wmma_f16(pf, load_b16_contig(&v_tile[t * 16 + nlo][half16 * 16]), acc[t]);
  }

  // ---- normalize and write f32 output [B, QL, H*DH] ----
  const int b = bh / NH, h = bh % NH;
  for (int r = 0; r < 8; r++) {
    const float inv = 1.0f / l_run[r];
    const int row_q = qt * 16 + r + 8 * half16;
    float* orow = out + ((size_t)b * QLEN + row_q) * (NH * DH) + h * DH;
    for (int t = 0; t < 8; t++) orow[t * 16 + nlo] = acc[t][r] * inv;
  }
}

// ---------------------------------------------------------------------------
extern "C" void kernel_launch(void* const* d_in, const int* in_sizes, int n_in,
                              void* d_out, int out_size, void* d_ws, size_t ws_size,
                              hipStream_t stream) {
  (void)in_sizes; (void)n_in; (void)out_size; (void)ws_size;
  const float* hs   = (const float*)d_in[0];
  const float* inp  = (const float*)d_in[1];
  const float* ln1g = (const float*)d_in[2];
  const float* ln1b = (const float*)d_in[3];
  const float* ln2g = (const float*)d_in[4];
  const float* ln2b = (const float*)d_in[5];
  const float* Wq   = (const float*)d_in[6];
  const float* bq   = (const float*)d_in[7];
  const float* Wk   = (const float*)d_in[8];
  const float* bk   = (const float*)d_in[9];
  const float* Wv   = (const float*)d_in[10];
  const float* bv   = (const float*)d_in[11];
  float* out = (float*)d_out;

  // ---- workspace carve-up (all f16, ~101 MB total) ----
  char* ws = (char*)d_ws;
  half_t* hs_n  = (half_t*)ws; ws += (size_t)BATCH * QLEN  * QDIM  * 2;   //  4 MB
  half_t* inp_n = (half_t*)ws; ws += (size_t)BATCH * KVLEN * KVDIM * 2;   // 24 MB
  half_t* WqT   = (half_t*)ws; ws += (size_t)QKCH * QDIM  * 2;            //  2 MB
  half_t* WkT   = (half_t*)ws; ws += (size_t)QKCH * KVDIM * 2;            //1.5 MB
  half_t* WvT   = (half_t*)ws; ws += (size_t)VCH  * KVDIM * 2;            //1.5 MB
  half_t* Qfb   = (half_t*)ws; ws += (size_t)BATCH * NH * QLEN  * DH * 2; //  4 MB
  half_t* Kfb   = (half_t*)ws; ws += (size_t)BATCH * NH * KVLEN * DH * 2; // 32 MB
  half_t* Vtb   = (half_t*)ws; ws += (size_t)BATCH * NH * DH * KVLEN * 2; // 32 MB

  // 1) LayerNorms -> f16
  ln_to_f16<QDIM ><<<BATCH * QLEN,  256, 0, stream>>>(hs,  ln1g, ln1b, hs_n);
  ln_to_f16<KVDIM><<<BATCH * KVLEN, 256, 0, stream>>>(inp, ln2g, ln2b, inp_n);

  // 2) weights -> transposed f16
  wtrans_f16<<<(QDIM  * QKCH + 255) / 256, 256, 0, stream>>>(Wq, WqT, QDIM,  QKCH);
  wtrans_f16<<<(KVDIM * QKCH + 255) / 256, 256, 0, stream>>>(Wk, WkT, KVDIM, QKCH);
  wtrans_f16<<<(KVDIM * VCH  + 255) / 256, 256, 0, stream>>>(Wv, WvT, KVDIM, VCH);

  // 3) projections (wave = 16x16 tile, 8 waves / 256-thread block)
  const float qscale = 1.0f / sqrtf((float)DH);
  {
    int tiles = ((BATCH * QLEN) / 16) * (QKCH / 16);                // 8192 waves
    proj_gemm<<<(tiles * 32 + 255) / 256, 256, 0, stream>>>(
        hs_n, WqT, bq, Qfb, BATCH * QLEN, QKCH, QDIM, QLEN, /*mode=*/0, qscale);
  }
  {
    int tiles = ((BATCH * KVLEN) / 16) * (QKCH / 16);               // 65536 waves
    proj_gemm<<<(tiles * 32 + 255) / 256, 256, 0, stream>>>(
        inp_n, WkT, bk, Kfb, BATCH * KVLEN, QKCH, KVDIM, KVLEN, /*mode=*/1, 1.0f);
  }
  {
    int tiles = ((BATCH * KVLEN) / 16) * (VCH / 16);
    proj_gemm<<<(tiles * 32 + 255) / 256, 256, 0, stream>>>(
        inp_n, WvT, bv, Vtb, BATCH * KVLEN, VCH, KVDIM, KVLEN, /*mode=*/2, 1.0f);
  }

  // 4) flash attention: 32 (b,h) x 8 blocks, 4 waves (q-tiles) per block
  flash_attn<<<(BATCH * NH) * 8, 128, 0, stream>>>(Qfb, Kfb, Vtb, out);
}
